// MultiLayerNeighborMessageFunction_2989297238773
// MI455X (gfx1250) — matrix operand order repaired and checked by reading, not compiled, not run on hardware
//
#include <hip/hip_runtime.h>

typedef _Float16 h16 __attribute__((ext_vector_type(16)));
typedef _Float16 h8  __attribute__((ext_vector_type(8)));
typedef _Float16 h4  __attribute__((ext_vector_type(4)));
typedef float    f8  __attribute__((ext_vector_type(8)));

#define NB      1024          // B
#define KNBR    20            // K
#define DMEM    256
#define DMSG    256
#define DRAW    512
#define MROWS   (NB + NB*KNBR)   // 21504 layer-1 rows

// ---------------- f32 -> f16 conversion ----------------
__global__ void cvt_f16_kernel(const float* __restrict__ src, _Float16* __restrict__ dst, int n) {
    int i = blockIdx.x * blockDim.x + threadIdx.x;
    if (i < n) dst[i] = (_Float16)src[i];
}

// ---------------- gather + K-sum (memory-bound stage, ~460MB @ 23.3TB/s) ----------------
// 64 threads per row, float4 loads => each row read is one coalesced 1KB burst.
// Prefetch next neighbor row to hide HBM latency on the random gather.
__global__ void gather_sum_kernel(const float4* __restrict__ mem4,
                                  const int* __restrict__ node_idx,
                                  const int* __restrict__ nbr_idx,
                                  h4* __restrict__ X1, h4* __restrict__ X2,
                                  int nrows) {
    int tid = blockIdx.x * 256 + threadIdx.x;
    int r = tid >> 6;          // row
    int t = tid & 63;          // float4 slot within row (64 * 16B = 1KB)
    if (r >= nrows) return;

    int node = node_idx[r];
    float4 v = mem4[(size_t)node * 64 + t];
    h4 hv; hv[0] = (_Float16)v.x; hv[1] = (_Float16)v.y; hv[2] = (_Float16)v.z; hv[3] = (_Float16)v.w;
    X1[(size_t)r * 64 + t] = hv;

    const int* nb = nbr_idx + (size_t)r * KNBR;
    float sx = 0.f, sy = 0.f, sz = 0.f, sw = 0.f;
    int j = nb[0];
    for (int k = 0; k < KNBR; ++k) {
        int jn = (k + 1 < KNBR) ? nb[k + 1] : j;
        __builtin_prefetch(&mem4[(size_t)jn * 64 + t], 0, 0);   // global_prefetch_b8
        float4 u = mem4[(size_t)j * 64 + t];
        sx += u.x; sy += u.y; sz += u.z; sw += u.w;
        j = jn;
    }
    h4 hs; hs[0] = (_Float16)sx; hs[1] = (_Float16)sy; hs[2] = (_Float16)sz; hs[3] = (_Float16)sw;
    X2[(size_t)r * 64 + t] = hs;
}

// ---------------- WMMA dual GEMM: relu(A1*W1^T + A2*W2^T + b1 + b2) ----------------
// One wave computes a 16(M) x 64(N) output tile: A loaded once per K-step,
// 4 B tiles / 4 wmma accumulators -> ~13 FLOP per L2 byte, 4x fewer A re-reads.
// A(16x32) layout used here: lanes 0-15 carry K 0..15, lanes 16-31 carry K 16..31 (matches B).
// B(32x16) layout used here: lane n<16 -> K 0..15 of col n; lane n>=16 -> K 16..31 of col n-16.
__device__ __forceinline__ void wmma_accum4(const _Float16* __restrict__ A,
                                            const _Float16* __restrict__ W,
                                            int Kd, int m, int nb0, int half,
                                            f8& c0, f8& c1, f8& c2, f8& c3) {
    const _Float16* pa  = A + (size_t)m * Kd + half * 16;         // 32B aligned
    const _Float16* pb0 = W + (size_t)nb0 * Kd + half * 16;       // 32B aligned
    const _Float16* pb1 = pb0 + (size_t)16 * Kd;
    const _Float16* pb2 = pb0 + (size_t)32 * Kd;
    const _Float16* pb3 = pb0 + (size_t)48 * Kd;
    for (int kb = 0; kb < Kd; kb += 32) {
        h16 a = *(const h16*)(pa + kb);
        h16 b0 = *(const h16*)(pb0 + kb);
        h16 b1 = *(const h16*)(pb1 + kb);
        h16 b2 = *(const h16*)(pb2 + kb);
        h16 b3 = *(const h16*)(pb3 + kb);
        c0 = __builtin_amdgcn_wmma_f32_16x16x32_f16(false, a, false, b0, (short)0, c0, false, false);
        c1 = __builtin_amdgcn_wmma_f32_16x16x32_f16(false, a, false, b1, (short)0, c1, false, false);
        c2 = __builtin_amdgcn_wmma_f32_16x16x32_f16(false, a, false, b2, (short)0, c2, false, false);
        c3 = __builtin_amdgcn_wmma_f32_16x16x32_f16(false, a, false, b3, (short)0, c3, false, false);
    }
}

__global__ void wmma_gemm_dual(const _Float16* __restrict__ A1, const _Float16* __restrict__ W1, int K1,
                               const _Float16* __restrict__ A2, const _Float16* __restrict__ W2, int K2,
                               const float* __restrict__ bias1, const float* __restrict__ bias2,
                               float* __restrict__ outF, _Float16* __restrict__ outH, int M) {
    int wid  = threadIdx.x >> 5;
    int lane = threadIdx.x & 31;
    int half = lane >> 4;
    int lrow = lane & 15;
    int wave_g = blockIdx.x * 8 + wid;       // 8 waves / 256-thread block
    int tile_m = wave_g >> 2;                // N = 256 -> 4 N-tiles of 64
    int tile_n = wave_g & 3;
    if (tile_m * 16 >= M) return;            // wave-uniform: EXEC stays all-1s

    int m   = tile_m * 16 + lrow;            // A row for this lane
    int nb0 = tile_n * 64 + lrow;            // first W row (= output column) for this lane

    f8 c0 = {}, c1 = {}, c2 = {}, c3 = {};
    wmma_accum4(A1, W1, K1, m, nb0, half, c0, c1, c2, c3);
    wmma_accum4(A2, W2, K2, m, nb0, half, c0, c1, c2, c3);

    f8* acc[4] = { &c0, &c1, &c2, &c3 };
    #pragma unroll
    for (int sub = 0; sub < 4; ++sub) {
        int n = tile_n * 64 + sub * 16 + lrow;
        float bv = bias1[n] + bias2[n];
        f8 c = *acc[sub];
        #pragma unroll
        for (int r = 0; r < 8; ++r) {
            int row = tile_m * 16 + r + half * 8;   // C/D VGPR layout
            float v = c[r] + bv;
            v = v > 0.f ? v : 0.f;                  // relu
            if (outF) outF[(size_t)row * 256 + n] = v;
            if (outH) outH[(size_t)row * 256 + n] = (_Float16)v;
        }
    }
}

// ---------------- neighbor aggregation over K rep1 rows ----------------
__global__ void agg_kernel(const _Float16* __restrict__ H1h, _Float16* __restrict__ AGGh) {
    int i = blockIdx.x * 256 + threadIdx.x;   // NB*256 threads
    int b = i >> 8, d = i & 255;
    const _Float16* p = H1h + ((size_t)NB + (size_t)b * KNBR) * 256 + d;
    float s = 0.f;
    #pragma unroll
    for (int k = 0; k < KNBR; ++k) s += (float)p[(size_t)k * 256];
    AGGh[(size_t)b * 256 + d] = (_Float16)s;
}

extern "C" void kernel_launch(void* const* d_in, const int* in_sizes, int n_in,
                              void* d_out, int out_size, void* d_ws, size_t ws_size,
                              hipStream_t stream) {
    const float* raw   = (const float*)d_in[0];
    const float* mem   = (const float*)d_in[1];
    const int*   src_n = (const int*)d_in[2];
    const int*   nl1   = (const int*)d_in[3];
    const int*   nl2   = (const int*)d_in[4];
    const float* W1_0  = (const float*)d_in[5];
    const float* b1_0  = (const float*)d_in[6];
    const float* W2_0  = (const float*)d_in[7];
    const float* b2_0  = (const float*)d_in[8];
    const float* W1_1  = (const float*)d_in[9];
    const float* b1_1  = (const float*)d_in[10];
    const float* W2_1  = (const float*)d_in[11];
    const float* b2_1  = (const float*)d_in[12];
    const float* msgW  = (const float*)d_in[13];
    const float* msgb  = (const float*)d_in[14];
    const float* aggW  = (const float*)d_in[15];
    const float* aggb  = (const float*)d_in[16];

    // carve workspace (all fully written before read; no cross-call state)
    char* ws = (char*)d_ws;
    auto carve = [&](size_t bytes) -> char* {
        char* p = ws; ws += (bytes + 255) & ~(size_t)255; return p;
    };
    _Float16* X1h   = (_Float16*)carve((size_t)MROWS * 256 * 2);
    _Float16* X2h   = (_Float16*)carve((size_t)MROWS * 256 * 2);
    _Float16* H1h   = (_Float16*)carve((size_t)MROWS * 256 * 2);
    _Float16* AGGh  = (_Float16*)carve((size_t)NB * 256 * 2);
    _Float16* REP2h = (_Float16*)carve((size_t)NB * 256 * 2);
    _Float16* RAWh  = (_Float16*)carve((size_t)NB * DRAW * 2);
    _Float16* W10h  = (_Float16*)carve((size_t)256 * 256 * 2);
    _Float16* W20h  = (_Float16*)carve((size_t)256 * 256 * 2);
    _Float16* W11h  = (_Float16*)carve((size_t)256 * 256 * 2);
    _Float16* W21h  = (_Float16*)carve((size_t)256 * 256 * 2);
    _Float16* AWh   = (_Float16*)carve((size_t)256 * 256 * 2);
    _Float16* MWh   = (_Float16*)carve((size_t)256 * DRAW * 2);

    // f16 conversions (weights + raw messages)
    cvt_f16_kernel<<<256, 256, 0, stream>>>(W1_0, W10h, 256 * 256);
    cvt_f16_kernel<<<256, 256, 0, stream>>>(W2_0, W20h, 256 * 256);
    cvt_f16_kernel<<<256, 256, 0, stream>>>(W1_1, W11h, 256 * 256);
    cvt_f16_kernel<<<256, 256, 0, stream>>>(W2_1, W21h, 256 * 256);
    cvt_f16_kernel<<<256, 256, 0, stream>>>(aggW, AWh,  256 * 256);
    cvt_f16_kernel<<<512, 256, 0, stream>>>(msgW, MWh,  256 * DRAW);
    cvt_f16_kernel<<<2048, 256, 0, stream>>>(raw, RAWh, NB * DRAW);

    const float4* mem4 = (const float4*)mem;
    // layer-1 gathers: source rows [0,B), neighbor rows [B, B+B*K)
    gather_sum_kernel<<<NB / 4, 256, 0, stream>>>(mem4, src_n, nl1,
                                                  (h4*)X1h, (h4*)X2h, NB);
    gather_sum_kernel<<<(NB * KNBR) / 4, 256, 0, stream>>>(mem4, nl1, nl2,
                                                  (h4*)(X1h + (size_t)NB * 256),
                                                  (h4*)(X2h + (size_t)NB * 256), NB * KNBR);

    // layer-1 GEMM: H1 = relu(X1*W1_0^T + X2*W2_0^T + b)
    // waves = (M/16)*4 N-tiles of 64 -> 5376 waves -> 672 blocks of 8 waves
    wmma_gemm_dual<<<(MROWS / 16) * 4 / 8, 256, 0, stream>>>(
        X1h, W10h, 256, X2h, W20h, 256, b1_0, b2_0, nullptr, H1h, MROWS);

    // aggregate 20 neighbor reps per batch element
    agg_kernel<<<NB, 256, 0, stream>>>(H1h, AGGh);

    // layer-2 GEMM: rep2 = relu(rep1_src*W1_1^T + agg*W2_1^T + b)  (256 waves -> 32 blocks)
    wmma_gemm_dual<<<(NB / 16) * 4 / 8, 256, 0, stream>>>(
        H1h, W11h, 256, AGGh, W21h, 256, b1_1, b2_1, nullptr, REP2h, NB);

    // output: relu(raw*msgW^T + rep2*aggW^T + b) -> f32 d_out
    wmma_gemm_dual<<<(NB / 16) * 4 / 8, 256, 0, stream>>>(
        RAWh, MWh, DRAW, REP2h, AWh, 256, msgb, aggb, (float*)d_out, nullptr, NB);
}
